// PointPnPNet_29557964931124
// MI455X (gfx1250) — matrix operand order repaired and checked by reading, not compiled, not run on hardware
//
#include <hip/hip_runtime.h>
#include <hip/hip_bf16.h>
#include <cstdint>
#include <cstddef>

typedef __attribute__((ext_vector_type(16))) _Float16 v16h;
typedef __attribute__((ext_vector_type(8)))  _Float16 v8h;
typedef __attribute__((ext_vector_type(8)))  float    v8f;
typedef __attribute__((ext_vector_type(4)))  float    v4f;
typedef __attribute__((ext_vector_type(4)))  unsigned int u32x4;
typedef __attribute__((ext_vector_type(4)))  int      i32x4;
typedef __attribute__((ext_vector_type(8)))  int      i32x8;

#define NEG_SLOPE 0.1f

__device__ __forceinline__ float leaky_f(float x) { return x >= 0.f ? x : NEG_SLOPE * x; }

// Problem constants
static constexpr int BS    = 32;
static constexpr int PTS   = 4096;          // 64*64
static constexpr int NPTS  = BS * PTS;      // 131072 total points
static constexpr int SPCHUNKS = 32;         // K-chunks for conv_sp GEMM
static constexpr int MXCHUNKS = 8;          // P-chunks for masked max pool

// ---------------------------------------------------------------------------
// Tensor Data Mover: 2-D tile (rows of tile0 8-byte units, tile1 rows,
// row stride stride0 in 8-byte units) from global -> LDS (packed rows).
// D# built per CDNA5 ISA 8.3/8.4: type=2, count=1, data_size=3 (8B),
// huge tensor extents so no OOB clipping occurs.
// ---------------------------------------------------------------------------
__device__ __forceinline__ void tdm_load_to_lds(unsigned lds_addr, unsigned long long gaddr,
                                                unsigned tile0, unsigned tile1,
                                                unsigned long long stride0) {
    u32x4 g0;
    g0[0] = 1u;                                            // count=1, user descriptor
    g0[1] = lds_addr;                                      // LDS byte address
    g0[2] = (unsigned)gaddr;                               // global addr [31:0]
    g0[3] = (unsigned)((gaddr >> 32) & 0x01FFFFFFu) | (2u << 30);  // [56:32] | type=2
    unsigned long long td0 = 1ull << 30;                   // tensor_dim0 (8B units)
    unsigned td1 = 1u << 20;                               // tensor_dim1
    i32x8 g1;
    g1[0] = (int)0x00030000u;                              // data_size=3 (8B), mask=0
    g1[1] = (int)((unsigned)(td0 & 0xFFFFull) << 16);      // [47:32]=0, td0[15:0]
    g1[2] = (int)((unsigned)((td0 >> 16) & 0xFFFFull) | ((td1 & 0xFFFFu) << 16));
    g1[3] = (int)(((td1 >> 16) & 0xFFFFu) | (tile0 << 16));// td1[31:16] | tile_dim0
    g1[4] = (int)tile1;                                    // tile_dim1 (tile_dim2=0)
    g1[5] = (int)(unsigned)stride0;                        // tensor_dim0_stride[31:0]
    g1[6] = (int)(unsigned)((stride0 >> 32) & 0xFFFFull);  // stride[47:32], dim1_stride=0
    g1[7] = 0;
    i32x4 gz; gz[0] = 0; gz[1] = 0; gz[2] = 0; gz[3] = 0;
#if defined(__clang_major__) && (__clang_major__ >= 23)
    i32x8 gz8; for (int i = 0; i < 8; ++i) gz8[i] = 0;
    __builtin_amdgcn_tensor_load_to_lds(g0, g1, gz, gz, gz8, 0);
#else
    __builtin_amdgcn_tensor_load_to_lds(g0, g1, gz, gz, 0);
#endif
}

__device__ __forceinline__ unsigned lds_off(const void* p) {
    return (unsigned)(size_t)p;   // low 32 bits of flat LDS address = LDS offset
}

// ---------------------------------------------------------------------------
// Pre-pack a 128x128 f32 weight (W[o][i]) into WMMA A-fragment order (f16).
// A-frag order used here, B loaded to match (16x32 f16): lane L (M=L&15, half=L>>4), element e: K = half*16 + e
// packed[((mt*4+ks)*32 + lane)*16 + e] = W[mt*16 + (lane&15)][ks*32 + K]
// ---------------------------------------------------------------------------
__global__ void prepack128_kernel(const float* __restrict__ w, _Float16* __restrict__ pa) {
    int t    = blockIdx.x * 256 + threadIdx.x;      // < 16384
    int e    = t & 15;
    int lane = (t >> 4) & 31;
    int kt   = t >> 9;                               // mt*4+ks, 0..31
    int mt   = kt >> 2, ks = kt & 3;
    int m    = mt * 16 + (lane & 15);
    int k    = ks * 32 + ((lane >> 4) << 4) + e;
    pa[t] = (_Float16)w[m * 128 + k];
}

// Pre-pack w_sp (128 o, 128 c, 128 k) as A matrix, M=128 (o), K=16384 (c*128+k).
__global__ void prepack_sp_kernel(const float* __restrict__ wsp, _Float16* __restrict__ pa) {
    int t    = blockIdx.x * 256 + threadIdx.x;      // < 2097152
    int e    = t & 15;
    int lane = (t >> 4) & 31;
    int rest = t >> 9;                               // mt*512 + ks, 0..4095
    int mt   = rest >> 9, ks = rest & 511;
    int m    = mt * 16 + (lane & 15);
    int kk   = ks * 32 + ((lane >> 4) << 4) + e;
    int c    = kk >> 7, k = kk & 127;
    pa[t] = (_Float16)wsp[(size_t)m * 16384 + c * 128 + k];
}

// ---------------------------------------------------------------------------
// Stage 1: xyz denorm + layer1 (5 -> 128) + leaky; output point-major f16.
// ---------------------------------------------------------------------------
__global__ void stage1_kernel(const float* __restrict__ coor, const float* __restrict__ extents,
                              const float* __restrict__ w1, const float* __restrict__ b1,
                              _Float16* __restrict__ h1t) {
    __shared__ float sw[128 * 5 + 128];
    for (int i = threadIdx.x; i < 128 * 5 + 128; i += 256)
        sw[i] = (i < 640) ? w1[i] : b1[i - 640];
    __syncthreads();

    int g = blockIdx.x * 256 + threadIdx.x;          // global point id, < 131072
    int b = g >> 12;
    int p = g & 4095;
    float xin[5];
    const float* cf = coor + (size_t)b * 5 * 4096 + p;
#pragma unroll
    for (int c = 0; c < 5; ++c) xin[c] = cf[c * 4096];
#pragma unroll
    for (int c = 0; c < 3; ++c) xin[c] = (xin[c] - 0.5f) * extents[b * 3 + c];

    _Float16* orow = h1t + (size_t)g * 128;
#pragma unroll
    for (int og = 0; og < 16; ++og) {
        v8h hv;
#pragma unroll
        for (int v = 0; v < 8; ++v) {
            int o = og * 8 + v;
            float acc = sw[640 + o];
#pragma unroll
            for (int i = 0; i < 5; ++i) acc += sw[o * 5 + i] * xin[i];
            hv[v] = (_Float16)leaky_f(acc);
        }
        *(v8h*)(orow + og * 8) = hv;
    }
}

// ---------------------------------------------------------------------------
// 128x128 x (128 x NPTS) GEMM via WMMA f16->f32.
// Packed A (32 KB) staged once per block into LDS via TDM; all 8 waves share.
// Each wave: 4 n-tiles (B frags in regs), A frags from LDS -> pure WMMA issue.
// grid: 256 blocks x 256 threads.
// ---------------------------------------------------------------------------
template <bool LEAKY, bool OUTF16>
__global__ void __launch_bounds__(256, 1)
gemm128_kernel(const _Float16* __restrict__ in, const _Float16* __restrict__ pa,
               const float* __restrict__ bias, void* __restrict__ outp) {
    __shared__ _Float16 As[16384];                   // 32 KB packed A
    int tid  = threadIdx.x;
    int wave = tid >> 5;
    int lane = tid & 31;
    int hf   = lane >> 4;
    int ntb  = (blockIdx.x * 8 + wave) * 4;          // first n-tile of this wave

    if (wave == 0) {
        tdm_load_to_lds(lds_off(As), (unsigned long long)(size_t)pa,
                        /*tile0(8B)=*/4096u, /*tile1=*/1u, /*stride0=*/4096ull);
        __builtin_amdgcn_s_wait_tensorcnt((short)0);
    }

    int row[4];
    v16h bf[4][4];
#pragma unroll
    for (int j = 0; j < 4; ++j) {
        row[j] = (ntb + j) * 16 + (lane & 15);
        const v16h* inrow = (const v16h*)(in + (size_t)row[j] * 128);
#pragma unroll
        for (int ks = 0; ks < 4; ++ks) bf[j][ks] = inrow[ks * 2 + hf];
    }
    __syncthreads();

    const v16h* Af = (const v16h*)As;
#pragma unroll
    for (int mt = 0; mt < 8; ++mt) {
        v16h af[4];
#pragma unroll
        for (int ks = 0; ks < 4; ++ks) af[ks] = Af[(mt * 4 + ks) * 32 + lane];

        int ob = mt * 16 + hf * 8;                   // 8 contiguous output channels
        v4f bz0 = *(const v4f*)(bias + ob);
        v4f bz1 = *(const v4f*)(bias + ob + 4);

#pragma unroll
        for (int j = 0; j < 4; ++j) {
            v8f acc = {};
#pragma unroll
            for (int ks = 0; ks < 4; ++ks)
                acc = __builtin_amdgcn_wmma_f32_16x16x32_f16(false, af[ks], false, bf[j][ks],
                                                             (short)0, acc, false, false);
            float r[8];
#pragma unroll
            for (int v = 0; v < 8; ++v) {
                float y = acc[v] + (v < 4 ? bz0[v] : bz1[v - 4]);
                if (LEAKY) y = leaky_f(y);
                r[v] = y;
            }
            if (OUTF16) {
                v8h hv;
#pragma unroll
                for (int v = 0; v < 8; ++v) hv[v] = (_Float16)r[v];
                *(v8h*)((_Float16*)outp + (size_t)row[j] * 128 + ob) = hv;
            } else {
                float* orow = (float*)outp + (size_t)row[j] * 128 + ob;
                v4f lo, hi;
#pragma unroll
                for (int v = 0; v < 4; ++v) { lo[v] = r[v]; hi[v] = r[v + 4]; }
                *(v4f*)orow = lo;
                *(v4f*)(orow + 4) = hi;
            }
        }
    }
}

// ---------------------------------------------------------------------------
// Region-masked max pool, P-chunked partials:
// pmx[chunk][(b*128+c)*8+m] = max over 512 points of h3[point][c]*region[b][m][p]
// grid: 256 blocks (b*8+chunk) x 128 threads (c).
// ---------------------------------------------------------------------------
__global__ void maxpool_partial_kernel(const float* __restrict__ h3, const float* __restrict__ region,
                                       float* __restrict__ pmx) {
    __shared__ float rg[8][256];
    int blk   = blockIdx.x;
    int b     = blk >> 3;
    int chunk = blk & 7;
    int t     = threadIdx.x;  // channel
    float mx[8];
#pragma unroll
    for (int m = 0; m < 8; ++m) mx[m] = -3.402823466e38f;

    int pbase = chunk * 512;
    for (int p0 = pbase; p0 < pbase + 512; p0 += 256) {
        for (int i = t; i < 8 * 256; i += 128) {
            int m = i >> 8, pp = i & 255;
            rg[m][pp] = region[(size_t)(b * 8 + m) * 4096 + p0 + pp];
        }
        __syncthreads();
        for (int pp = 0; pp < 256; ++pp) {
            float val = h3[((size_t)(b * 4096 + p0 + pp)) * 128 + t];
#pragma unroll
            for (int m = 0; m < 8; ++m) mx[m] = fmaxf(mx[m], val * rg[m][pp]);
        }
        __syncthreads();
    }
#pragma unroll
    for (int m = 0; m < 8; ++m)
        pmx[(size_t)chunk * 32768 + (size_t)(b * 128 + t) * 8 + m] = mx[m];
}

// Reduce partial maxima over chunks: x5[t] = max_chunk pmx[chunk][t]
__global__ void maxpool_reduce_kernel(const float* __restrict__ pmx, float* __restrict__ x5) {
    int t = blockIdx.x * 256 + threadIdx.x;          // < 32768
    float mx = -3.402823466e38f;
#pragma unroll
    for (int c = 0; c < MXCHUNKS; ++c) mx = fmaxf(mx, pmx[(size_t)c * 32768 + t]);
    x5[t] = mx;
}

// ---------------------------------------------------------------------------
// Softmax over channels (per b,m) + stable descending 8-sort per (b,f).
// 32 blocks x 128 threads (thread = channel).
// ---------------------------------------------------------------------------
__global__ void softmax_sort_kernel(const float* __restrict__ x5, float* __restrict__ xs,
                                    int* __restrict__ idx) {
    __shared__ float red[128];
    int b = blockIdx.x;
    int c = threadIdx.x;
    float v[8], e[8];
#pragma unroll
    for (int m = 0; m < 8; ++m) v[m] = x5[(size_t)(b * 128 + c) * 8 + m];

    for (int m = 0; m < 8; ++m) {
        red[c] = v[m];
        __syncthreads();
        for (int s = 64; s > 0; s >>= 1) { if (c < s) red[c] = fmaxf(red[c], red[c + s]); __syncthreads(); }
        float mxv = red[0];
        __syncthreads();
        float ev = __expf(v[m] - mxv);
        red[c] = ev;
        __syncthreads();
        for (int s = 64; s > 0; s >>= 1) { if (c < s) red[c] += red[c + s]; __syncthreads(); }
        float sm = red[0];
        __syncthreads();
        e[m] = ev / sm;
    }
#pragma unroll
    for (int m = 0; m < 8; ++m) xs[(size_t)(b * 128 + c) * 8 + m] = e[m];

    // stable descending selection sort (matches lax.top_k tie-breaking)
    int used = 0;
    for (int n = 0; n < 8; ++n) {
        int best = 0;
        float bv = -3.402823466e38f;
        for (int m = 0; m < 8; ++m) {
            if (!((used >> m) & 1) && e[m] > bv) { bv = e[m]; best = m; }
        }
        used |= 1 << best;
        idx[(size_t)(b * 128 + c) * 8 + n] = best;
    }
}

// ---------------------------------------------------------------------------
// Materialize sp columns (f16): spT[(b*8+r)*16384 + c*128 + k] = xs[b,c,idx[b,k,r]]
// ---------------------------------------------------------------------------
__global__ void build_spT_kernel(const float* __restrict__ xs, const int* __restrict__ idx,
                                 _Float16* __restrict__ spT) {
    int t  = blockIdx.x * 256 + threadIdx.x;   // < 4194304
    int kk = t & 16383;
    int n  = t >> 14;
    int b  = n >> 3, r = n & 7;
    int c  = kk >> 7, k = kk & 127;
    int id = idx[(size_t)(b * 128 + k) * 8 + r];
    spT[t] = (_Float16)xs[(size_t)(b * 128 + c) * 8 + id];
}

// ---------------------------------------------------------------------------
// conv_sp GEMM, K-chunked: M=128 (o), K=16384, N=256 (b*8+r).
// Block = one K-chunk (16 k-steps) shared by 8 waves (8 n-tiles).
// A staged per 4-k-step group via TDM into 32 KB LDS; WMMAs fed from LDS.
// grid: 64 blocks (chunk*2 + ntgroup) x 256 threads.
// psp[chunk][col*128 + o] = partial sum.
// ---------------------------------------------------------------------------
__global__ void __launch_bounds__(256, 1)
gemm_sp_kernel(const _Float16* __restrict__ spT, const _Float16* __restrict__ pa,
               float* __restrict__ psp) {
    __shared__ _Float16 As[16384];               // 32 KB: 8 m-tiles x 4 k-steps
    int tid   = threadIdx.x;
    int wave  = tid >> 5;
    int lane  = tid & 31;
    int hf    = lane >> 4;
    int chunk = blockIdx.x >> 1;                 // 0..31
    int ntg   = blockIdx.x & 1;                  // 0..1
    int nt    = ntg * 8 + wave;                  // 0..15
    int col   = nt * 16 + (lane & 15);           // 0..255

    v8f acc[8];
#pragma unroll
    for (int mt = 0; mt < 8; ++mt) acc[mt] = (v8f){};

    const v16h* sprow = (const v16h*)(spT + (size_t)col * 16384);
    const v16h* Af    = (const v16h*)As;
    int ks0 = chunk * 16;
    for (int g = 0; g < 4; ++g) {
        __syncthreads();                         // previous group's LDS reads done
        if (wave == 0) {
            // 8 rows (m-tiles) x 4 KB, row stride 512 KB, packed into LDS
            tdm_load_to_lds(lds_off(As),
                            (unsigned long long)(size_t)pa + (size_t)(ks0 + g * 4) * 1024,
                            /*tile0(8B)=*/512u, /*tile1=*/8u, /*stride0(8B)=*/65536ull);
            __builtin_amdgcn_s_wait_tensorcnt((short)0);
        }
        __syncthreads();
#pragma unroll
        for (int ksl = 0; ksl < 4; ++ksl) {
            v16h bf = sprow[(ks0 + g * 4 + ksl) * 2 + hf];
#pragma unroll
            for (int mt = 0; mt < 8; ++mt) {
                v16h af = Af[mt * 128 + ksl * 32 + lane];
                acc[mt] = __builtin_amdgcn_wmma_f32_16x16x32_f16(false, af, false, bf,
                                                                 (short)0, acc[mt], false, false);
            }
        }
    }
    float* prow = psp + (size_t)chunk * 32768 + (size_t)col * 128;
#pragma unroll
    for (int mt = 0; mt < 8; ++mt) {
        int ob = mt * 16 + hf * 8;
        v4f lo, hi;
#pragma unroll
        for (int v = 0; v < 4; ++v) { lo[v] = acc[mt][v]; hi[v] = acc[mt][v + 4]; }
        *(v4f*)(prow + ob)     = lo;
        *(v4f*)(prow + ob + 4) = hi;
    }
}

// Reduce conv_sp partials over chunks, add bias, leaky, scatter to (B,1024).
__global__ void gemm_sp_reduce_kernel(const float* __restrict__ psp, const float* __restrict__ bsp,
                                      float* __restrict__ xflat) {
    int t = blockIdx.x * 256 + threadIdx.x;          // < 32768
    int col = t >> 7, o = t & 127;
    float s = 0.f;
#pragma unroll
    for (int c = 0; c < SPCHUNKS; ++c) s += psp[(size_t)c * 32768 + t];
    float y = leaky_f(s + bsp[o]);
    int b = col >> 3, r = col & 7;
    xflat[(size_t)b * 1024 + o * 8 + r] = y;
}

// ---------------------------------------------------------------------------
// Fully-connected: out[b][j] = act(bias[j] + sum_i in[b][i]*w[j][i])
// ---------------------------------------------------------------------------
__global__ void fc_kernel(const float* __restrict__ in, const float* __restrict__ w,
                          const float* __restrict__ bias, float* __restrict__ out,
                          int In, int Out, int do_leaky) {
    int g = blockIdx.x * blockDim.x + threadIdx.x;
    int b = g / Out, j = g - b * Out;
    const float* xr = in + (size_t)b * In;
    const float* wr = w + (size_t)j * In;
    float acc = bias[j];
    for (int i = 0; i < In; i += 4) {
        v4f xv = *(const v4f*)(xr + i);
        v4f wv = *(const v4f*)(wr + i);
        acc += xv[0] * wv[0] + xv[1] * wv[1] + xv[2] * wv[2] + xv[3] * wv[3];
    }
    out[g] = do_leaky ? leaky_f(acc) : acc;
}

// Heads: rot (32,4) then t (32,3), concatenated flat into d_out.
__global__ void heads_kernel(const float* __restrict__ x, const float* __restrict__ fcrw,
                             const float* __restrict__ fcrb, const float* __restrict__ fctw,
                             const float* __restrict__ fctb, float* __restrict__ out) {
    int t = threadIdx.x;
    if (t >= 224) return;
    int b = t / 7, j = t % 7;
    const float* xr = x + (size_t)b * 256;
    if (j < 4) {
        const float* wr = fcrw + (size_t)j * 256;
        float acc = fcrb[j];
        for (int i = 0; i < 256; ++i) acc += xr[i] * wr[i];
        out[b * 4 + j] = acc;
    } else {
        int jj = j - 4;
        const float* wr = fctw + (size_t)jj * 256;
        float acc = fctb[jj];
        for (int i = 0; i < 256; ++i) acc += xr[i] * wr[i];
        out[128 + b * 3 + jj] = acc;
    }
}

// ---------------------------------------------------------------------------
extern "C" void kernel_launch(void* const* d_in, const int* in_sizes, int n_in,
                              void* d_out, int out_size, void* d_ws, size_t ws_size,
                              hipStream_t stream) {
    const float* coor    = (const float*)d_in[0];
    const float* region  = (const float*)d_in[1];
    const float* extents = (const float*)d_in[2];
    const float* w1      = (const float*)d_in[3];
    const float* b1      = (const float*)d_in[4];
    const float* w2      = (const float*)d_in[5];
    const float* b2      = (const float*)d_in[6];
    const float* w3      = (const float*)d_in[7];
    const float* b3      = (const float*)d_in[8];
    const float* w_sp    = (const float*)d_in[9];
    const float* b_sp    = (const float*)d_in[10];
    const float* fc1w    = (const float*)d_in[11];
    const float* fc1b    = (const float*)d_in[12];
    const float* fc2w    = (const float*)d_in[13];
    const float* fc2b    = (const float*)d_in[14];
    const float* fcrw    = (const float*)d_in[15];
    const float* fcrb    = (const float*)d_in[16];
    const float* fctw    = (const float*)d_in[17];
    const float* fctb    = (const float*)d_in[18];
    float* out = (float*)d_out;

    char* wp = (char*)d_ws;
    auto carve = [&](size_t bytes) -> void* {
        void* p = (void*)wp;
        wp += (bytes + 255) & ~(size_t)255;
        return p;
    };
    _Float16* paW2  = (_Float16*)carve((size_t)128 * 128 * 2);
    _Float16* paW3  = (_Float16*)carve((size_t)128 * 128 * 2);
    _Float16* paWsp = (_Float16*)carve((size_t)128 * 16384 * 2);
    _Float16* h1t   = (_Float16*)carve((size_t)NPTS * 128 * 2);
    _Float16* h2t   = (_Float16*)carve((size_t)NPTS * 128 * 2);
    float*    h3t   = (float*)carve((size_t)NPTS * 128 * 4);
    float*    pmx   = (float*)carve((size_t)MXCHUNKS * 32768 * 4);
    float*    x5    = (float*)carve((size_t)BS * 128 * 8 * 4);
    float*    xs    = (float*)carve((size_t)BS * 128 * 8 * 4);
    int*      idx   = (int*)carve((size_t)BS * 128 * 8 * 4);
    _Float16* spT   = (_Float16*)carve((size_t)256 * 16384 * 2);
    float*    psp   = (float*)carve((size_t)SPCHUNKS * 32768 * 4);
    float*    xflat = (float*)carve((size_t)BS * 1024 * 4);
    float*    fc1o  = (float*)carve((size_t)BS * 512 * 4);
    float*    fc2o  = (float*)carve((size_t)BS * 256 * 4);

    // weight pre-pack into WMMA A-fragment order
    prepack128_kernel<<<64, 256, 0, stream>>>(w2, paW2);
    prepack128_kernel<<<64, 256, 0, stream>>>(w3, paW3);
    prepack_sp_kernel<<<8192, 256, 0, stream>>>(w_sp, paWsp);

    // stage 1: denorm + layer1 + leaky -> f16 point-major
    stage1_kernel<<<512, 256, 0, stream>>>(coor, extents, w1, b1, h1t);

    // layers 2 & 3 via WMMA (A staged in LDS via TDM, 4 n-tiles per wave)
    gemm128_kernel<true, true><<<256, 256, 0, stream>>>(h1t, paW2, b2, (void*)h2t);
    gemm128_kernel<false, false><<<256, 256, 0, stream>>>(h2t, paW3, b3, (void*)h3t);

    // region-masked max pool (chunked + reduce)
    maxpool_partial_kernel<<<256, 128, 0, stream>>>(h3t, region, pmx);
    maxpool_reduce_kernel<<<128, 256, 0, stream>>>(pmx, x5);

    // softmax over channels + per-channel region sort
    softmax_sort_kernel<<<32, 128, 0, stream>>>(x5, xs, idx);

    // build sp columns + conv_sp WMMA GEMM (K-chunked, TDM-staged A) + reduce
    build_spT_kernel<<<16384, 256, 0, stream>>>(xs, idx, spT);
    gemm_sp_kernel<<<64, 256, 0, stream>>>(spT, paWsp, psp);
    gemm_sp_reduce_kernel<<<128, 256, 0, stream>>>(psp, b_sp, xflat);

    // FC stack + heads
    fc_kernel<<<64, 256, 0, stream>>>(xflat, fc1w, fc1b, fc1o, 1024, 512, 1);
    fc_kernel<<<32, 256, 0, stream>>>(fc1o, fc2w, fc2b, fc2o, 512, 256, 1);
    heads_kernel<<<1, 256, 0, stream>>>(fc2o, fcrw, fcrb, fctw, fctb, out);
}
